// C3Dnet_34600256537193
// MI455X (gfx1250) — compile-verified, matrix operand order checked by reading, not run on hardware
//
#include <hip/hip_runtime.h>
#include <hip/hip_bf16.h>
#include <hip/hip_fp16.h>

typedef __attribute__((ext_vector_type(16))) _Float16 v16h;
typedef __attribute__((ext_vector_type(8)))  float    v8f;
typedef uint32_t u32x4 __attribute__((ext_vector_type(4)));
typedef uint32_t u32x8 __attribute__((ext_vector_type(8)));
typedef _Float16 half_t;

// ---------------------------------------------------------------------------
// Tensor Data Mover: DMA a 2D tile (rows x cols f16, row stride == cols) from
// global memory into LDS.  D# per CDNA5 ISA ch.8: group0 = {count/type, lds
// addr, 57b global addr}, group1 = {data_size=2B, tensor dims, tile dims,
// strides}.  Issued by one wave; completion via TENSORcnt.
// ---------------------------------------------------------------------------
__device__ __forceinline__ void tdm_load_tile_f16(const half_t* gsrc, uint32_t lds_off,
                                                  uint32_t rows, uint32_t cols) {
  uint64_t ga = (uint64_t)(uintptr_t)gsrc;
  u32x4 g0;
  g0[0] = 1u;                                               // count=1 (valid), user mode
  g0[1] = lds_off;                                          // lds_addr (bytes)
  g0[2] = (uint32_t)ga;                                     // global_addr[31:0]
  g0[3] = ((uint32_t)(ga >> 32) & 0x01FFFFFFu) | (2u << 30);// global_addr[56:32] | type=2
  u32x8 g1;
  g1[0] = 1u << 16;       // workgroup_mask=0, data_size=1 (2 bytes)
  g1[1] = cols << 16;     // tensor_dim0[15:0] at bits 63:48
  g1[2] = rows << 16;     // tensor_dim0[31:16]=0 | tensor_dim1[15:0] at bits 95:80
  g1[3] = cols << 16;     // tensor_dim1 hi = 0 | tile_dim0 at bits 127:112
  g1[4] = rows;           // tile_dim1 at bits 143:128, tile_dim2=0
  g1[5] = cols;           // tensor_dim0_stride[31:0] at bits 191:160
  g1[6] = 0;              // tensor_dim0_stride hi, tensor_dim1_stride lo
  g1[7] = 0;
  asm volatile("tensor_load_to_lds %0, %1" :: "s"(g0), "s"(g1) : "memory");
}

// ---------------------------------------------------------------------------
// Weight f32 -> f16 (with optional cin zero-padding), row-major [co][cin_pad]
// ---------------------------------------------------------------------------
__global__ void cvt_w_kernel(const float* __restrict__ w, half_t* __restrict__ o,
                             int co, int cin, int cin_pad) {
  int i = blockIdx.x * 256 + threadIdx.x;
  int total = co * cin_pad;
  if (i >= total) return;
  int r = i / cin_pad, c = i - r * cin_pad;
  float v = (c < cin) ? w[r * cin + c] : 0.f;
  o[i] = (half_t)v;
}

// xyz [B][N][3] f32 -> [B][N][32] f16 zero-padded (point-major, ch contiguous)
__global__ void xyz_pad_kernel(const float* __restrict__ xyz, half_t* __restrict__ o,
                               int BN) {
  int i = blockIdx.x * 256 + threadIdx.x;
  if (i >= BN * 32) return;
  int n = i >> 5, c = i & 31;
  o[i] = (c < 3) ? (half_t)xyz[n * 3 + c] : (half_t)0.f;
}

// ---------------------------------------------------------------------------
// _random_sample: out[b][n'][c] = max_{j<16} in[b][pool[n',j]][c]   (f16)
// ---------------------------------------------------------------------------
__global__ void pool_max_kernel(const half_t* __restrict__ in, const int* __restrict__ idx,
                                half_t* __restrict__ out, int B, int Nin, int Nout, int C) {
  int i = blockIdx.x * 256 + threadIdx.x;
  int total = B * Nout * C;
  if (i >= total) return;
  int c = i % C;
  int n = (i / C) % Nout;
  int b = i / (C * Nout);
  const int* ip = idx + ((size_t)b * Nout + n) * 16;
  float m = -1e30f;
  #pragma unroll 4
  for (int j = 0; j < 16; ++j) {
    int r = ip[j];
    m = fmaxf(m, (float)in[((size_t)b * Nin + r) * C + c]);
  }
  out[i] = (half_t)m;
}

// ---------------------------------------------------------------------------
// Fused 1x1 conv + BN + ReLU via WMMA.  Input = concat(src0[CIN0], src1[CIN1])
// with optional per-point gather index per source (fuses _interp).
// One wave tile = 16 out-channels x 16 points. 8 waves/block = 128 points.
// Weight tile (16 x CIN) staged in LDS by the Tensor Data Mover.
// ---------------------------------------------------------------------------
template <int CIN0, int CIN1, bool F32OUT>
__global__ __launch_bounds__(256)
void conv1x1_wmma(const half_t* __restrict__ src0, const int* __restrict__ idx0, int nsrc0,
                  const half_t* __restrict__ src1, const int* __restrict__ idx1, int nsrc1,
                  const half_t* __restrict__ W,
                  const float* __restrict__ gam, const float* __restrict__ bet,
                  half_t* __restrict__ outh, float* __restrict__ outf,
                  int N, int co) {
  constexpr int CIN = CIN0 + CIN1;
  __shared__ half_t wlds[16 * CIN];
  const int m0 = blockIdx.x * 16;
  if (threadIdx.x < 32) {  // wave 0 issues the TDM DMA for the weight tile
    tdm_load_tile_f16(W + (size_t)m0 * CIN, (uint32_t)(uintptr_t)wlds, 16u, (uint32_t)CIN);
    __builtin_amdgcn_s_wait_tensorcnt(0);
  }
  __syncthreads();

  const int lane  = threadIdx.x & 31;
  const int wid   = threadIdx.x >> 5;
  const int col   = lane & 15;   // WMMA N column = point
  const int khalf = lane >> 4;   // K half-select: lane half h holds k = 16h..16h+15 (contiguous order)
  const int b     = blockIdx.z;

  int pt_raw = blockIdx.y * 128 + wid * 16 + col;
  int pt = (pt_raw < N) ? pt_raw : (N - 1);  // clamp: WMMA needs full EXEC

  int row0 = idx0 ? idx0[(size_t)b * N + pt] : pt;
  const half_t* p0 = src0 + ((size_t)b * nsrc0 + row0) * CIN0 + khalf * 16;
  const half_t* p1 = nullptr;
  if (CIN1 > 0) {
    int row1 = idx1 ? idx1[(size_t)b * N + pt] : pt;
    p1 = src1 + ((size_t)b * nsrc1 + row1) * CIN1 + khalf * 16;
  }
  const half_t* wp = wlds + col * CIN + khalf * 16;

  v8f acc = {};
  #pragma unroll
  for (int k0 = 0; k0 < CIN; k0 += 32) {
    v16h a = *(const v16h*)(wp + k0);                        // ds_load_b128 x2
    v16h bm;
    if (k0 < CIN0) bm = *(const v16h*)(p0 + k0);             // global_load_b128 x2
    else           bm = *(const v16h*)(p1 + (k0 - CIN0));
    acc = __builtin_amdgcn_wmma_f32_16x16x32_f16(false, a, false, bm,
                                                 (short)0, acc, false, false);
  }

  const int chbase = m0 + khalf * 8;  // D layout: VGPR g -> M = g + 8*khalf
  if (pt_raw < N) {
    if constexpr (F32OUT) {
      float* o = outf + ((size_t)b * co + chbase) * N + pt;  // [b][c][n]
      #pragma unroll
      for (int g = 0; g < 8; ++g)
        o[(size_t)g * N] = fmaxf(acc[g] * gam[chbase + g] + bet[chbase + g], 0.f);
    } else {
      alignas(16) half_t tmp[8];
      #pragma unroll
      for (int g = 0; g < 8; ++g)
        tmp[g] = (half_t)fmaxf(acc[g] * gam[chbase + g] + bet[chbase + g], 0.f);
      *(uint4*)(outh + ((size_t)b * N + pt) * co + chbase) = *(uint4*)tmp;
    }
  }
}

// ---------------------------------------------------------------------------
// Fused LFA: gather k=16 neighbors + concat(center, neighbor) + 1x1 conv
// + BN + ReLU + max over k.  WMMA N dimension == the 16 neighbors, so the
// max-over-k is a 4-step shfl_xor butterfly on the accumulator fragment.
// One wave tile = 16 out-channels x 1 point; each wave loops over 4 points.
// For CIN <= 256 the A fragments are cached in registers across the 4 points.
// ---------------------------------------------------------------------------
template <int D>
__global__ __launch_bounds__(256)
void lfa_wmma(const half_t* __restrict__ f, const int* __restrict__ neigh,
              const half_t* __restrict__ W,
              const float* __restrict__ gam, const float* __restrict__ bet,
              half_t* __restrict__ out, int N, int co) {
  constexpr int CIN   = 2 * D;
  constexpr int KFRAG = CIN / 32;
  constexpr bool AREG = (KFRAG <= 8);   // <= 64 VGPRs of cached A
  __shared__ half_t wlds[16 * CIN];
  const int m0 = blockIdx.x * 16;
  if (threadIdx.x < 32) {
    tdm_load_tile_f16(W + (size_t)m0 * CIN, (uint32_t)(uintptr_t)wlds, 16u, (uint32_t)CIN);
    __builtin_amdgcn_s_wait_tensorcnt(0);
  }
  __syncthreads();

  const int lane  = threadIdx.x & 31;
  const int wid   = threadIdx.x >> 5;
  const int col   = lane & 15;   // neighbor index j (WMMA column)
  const int khalf = lane >> 4;
  const int b     = blockIdx.z;
  const half_t* wp = wlds + col * CIN + khalf * 16;
  const half_t* fb = f + (size_t)b * N * D;

  v16h areg[AREG ? KFRAG : 1];
  if constexpr (AREG) {
    #pragma unroll
    for (int kf = 0; kf < KFRAG; ++kf) areg[kf] = *(const v16h*)(wp + kf * 32);
  }

  for (int pp = 0; pp < 4; ++pp) {
    const int pt = blockIdx.y * 32 + wid * 4 + pp;  // N % 32 == 0 for all layers
    const int nb = neigh[((size_t)b * N + pt) * 16 + col];
    const half_t* pc = fb + (size_t)pt * D + khalf * 16;  // center rows  [0,D)
    const half_t* pn = fb + (size_t)nb * D + khalf * 16;  // neighbor rows [D,2D)

    v8f acc = {};
    #pragma unroll
    for (int k0 = 0; k0 < CIN; k0 += 32) {
      v16h a;
      if constexpr (AREG) a = areg[k0 / 32];
      else                a = *(const v16h*)(wp + k0);
      v16h bm = (k0 < D) ? *(const v16h*)(pc + k0)
                         : *(const v16h*)(pn + (k0 - D));
      acc = __builtin_amdgcn_wmma_f32_16x16x32_f16(false, a, false, bm,
                                                   (short)0, acc, false, false);
    }

    // max over the 16 neighbor columns (lanes within each 16-lane half)
    #pragma unroll
    for (int g = 0; g < 8; ++g) {
      float v = acc[g];
      v = fmaxf(v, __shfl_xor(v, 1, 32));
      v = fmaxf(v, __shfl_xor(v, 2, 32));
      v = fmaxf(v, __shfl_xor(v, 4, 32));
      v = fmaxf(v, __shfl_xor(v, 8, 32));
      acc[g] = v;
    }

    const int chbase = m0 + khalf * 8;
    if (col == 0) {  // lanes 0 and 16 hold rows 0..7 / 8..15 after butterfly
      alignas(16) half_t tmp[8];
      #pragma unroll
      for (int g = 0; g < 8; ++g)
        tmp[g] = (half_t)fmaxf(acc[g] * gam[chbase + g] + bet[chbase + g], 0.f);
      *(uint4*)(out + ((size_t)b * N + pt) * co + chbase) = *(uint4*)tmp;
    }
  }
}

// ---------------------------------------------------------------------------
// Host orchestration
// ---------------------------------------------------------------------------
extern "C" void kernel_launch(void* const* d_in, const int* in_sizes, int n_in,
                              void* d_out, int out_size, void* d_ws, size_t ws_size,
                              hipStream_t stream) {
  (void)in_sizes; (void)n_in; (void)out_size; (void)ws_size;
  const int B = 2;
  const int N0 = 16384, N1 = 4096, N2 = 1024, N3 = 256, N4 = 64;

  const float* xyz    = (const float*)d_in[0];
  const int*   neigh0 = (const int*)d_in[1];
  const int*   pool0  = (const int*)d_in[2];
  const int*   up0    = (const int*)d_in[3];
  const int*   neigh1 = (const int*)d_in[4];
  const int*   pool1  = (const int*)d_in[5];
  const int*   up1    = (const int*)d_in[6];
  const int*   neigh2 = (const int*)d_in[7];
  const int*   pool2  = (const int*)d_in[8];
  const int*   up2    = (const int*)d_in[9];
  const int*   neigh3 = (const int*)d_in[10];
  const int*   pool3  = (const int*)d_in[11];
  const int*   up3    = (const int*)d_in[12];
  auto W = [&](int i) { return (const float*)d_in[i]; };

  // workspace bump allocator (256B aligned)
  char* wsb = (char*)d_ws;
  size_t off = 0;
  auto carve = [&](size_t bytes) -> char* {
    char* p = wsb + off;
    off += (bytes + 255) & ~(size_t)255;
    return p;
  };
  auto carveh = [&](size_t halves) { return (half_t*)carve(halves * sizeof(half_t)); };

  // f16 weights (fc0 padded 3 -> 32)
  half_t* fc0w = carveh((size_t)64 * 32);
  half_t* l1w  = carveh((size_t)128 * 128);
  half_t* l2w  = carveh((size_t)256 * 256);
  half_t* l3w  = carveh((size_t)512 * 512);
  half_t* l4w  = carveh((size_t)1024 * 1024);
  half_t* dtw  = carveh((size_t)1024 * 1024);
  half_t* d4w  = carveh((size_t)512 * 2048);
  half_t* d3w  = carveh((size_t)256 * 1024);
  half_t* d2w  = carveh((size_t)128 * 512);
  half_t* d1w  = carveh((size_t)64 * 256);
  half_t* d0w  = carveh((size_t)64 * 64);

  // f16 activations, layout [B][N][C]
  half_t* xyzp = carveh((size_t)B * N0 * 32);
  half_t* a0   = carveh((size_t)B * N0 * 64);
  half_t* o1   = carveh((size_t)B * N0 * 128);
  half_t* o1p  = carveh((size_t)B * N1 * 128);
  half_t* o2   = carveh((size_t)B * N1 * 256);
  half_t* o2p  = carveh((size_t)B * N2 * 256);
  half_t* o3   = carveh((size_t)B * N2 * 512);
  half_t* o3p  = carveh((size_t)B * N3 * 512);
  half_t* o4   = carveh((size_t)B * N3 * 1024);
  half_t* o4p  = carveh((size_t)B * N4 * 1024);
  half_t* dto  = carveh((size_t)B * N4 * 1024);
  half_t* d4o  = carveh((size_t)B * N4 * 512);
  half_t* d3o  = carveh((size_t)B * N3 * 256);
  half_t* d2o  = carveh((size_t)B * N2 * 128);
  half_t* d1o  = carveh((size_t)B * N1 * 64);

  auto cvt = [&](const float* w, half_t* o, int co, int cin, int cinp) {
    int tot = co * cinp;
    cvt_w_kernel<<<dim3((tot + 255) / 256), dim3(256), 0, stream>>>(w, o, co, cin, cinp);
  };
  cvt(W(13), fc0w, 64, 3, 32);
  cvt(W(16), l1w, 128, 128, 128);
  cvt(W(19), l2w, 256, 256, 256);
  cvt(W(22), l3w, 512, 512, 512);
  cvt(W(25), l4w, 1024, 1024, 1024);
  cvt(W(28), dtw, 1024, 1024, 1024);
  cvt(W(31), d4w, 512, 2048, 2048);
  cvt(W(34), d3w, 256, 1024, 1024);
  cvt(W(37), d2w, 128, 512, 512);
  cvt(W(40), d1w, 64, 256, 256);
  cvt(W(43), d0w, 64, 64, 64);

  xyz_pad_kernel<<<dim3((B * N0 * 32 + 255) / 256), dim3(256), 0, stream>>>(xyz, xyzp, B * N0);

  // ---------------- encoder ----------------
  // fc0: 32(pad) -> 64 @ N0
  conv1x1_wmma<32, 0, false><<<dim3(64 / 16, N0 / 128, B), 256, 0, stream>>>(
      xyzp, nullptr, N0, nullptr, nullptr, 0, fc0w, W(14), W(15), a0, nullptr, N0, 64);
  // l1 LFA: d=64 -> 128 @ N0
  lfa_wmma<64><<<dim3(128 / 16, N0 / 32, B), 256, 0, stream>>>(
      a0, neigh0, l1w, W(17), W(18), o1, N0, 128);
  pool_max_kernel<<<dim3((B * N1 * 128 + 255) / 256), 256, 0, stream>>>(
      o1, pool0, o1p, B, N0, N1, 128);
  // l2 LFA: d=128 -> 256 @ N1
  lfa_wmma<128><<<dim3(256 / 16, N1 / 32, B), 256, 0, stream>>>(
      o1p, neigh1, l2w, W(20), W(21), o2, N1, 256);
  pool_max_kernel<<<dim3((B * N2 * 256 + 255) / 256), 256, 0, stream>>>(
      o2, pool1, o2p, B, N1, N2, 256);
  // l3 LFA: d=256 -> 512 @ N2
  lfa_wmma<256><<<dim3(512 / 16, N2 / 32, B), 256, 0, stream>>>(
      o2p, neigh2, l3w, W(23), W(24), o3, N2, 512);
  pool_max_kernel<<<dim3((B * N3 * 512 + 255) / 256), 256, 0, stream>>>(
      o3, pool2, o3p, B, N2, N3, 512);
  // l4 LFA: d=512 -> 1024 @ N3
  lfa_wmma<512><<<dim3(1024 / 16, N3 / 32, B), 256, 0, stream>>>(
      o3p, neigh3, l4w, W(26), W(27), o4, N3, 1024);
  pool_max_kernel<<<dim3((B * N4 * 1024 + 255) / 256), 256, 0, stream>>>(
      o4, pool3, o4p, B, N3, N4, 1024);

  // ---------------- decoder ----------------
  // dt: 1024 -> 1024 @ N4
  conv1x1_wmma<1024, 0, false><<<dim3(1024 / 16, 1, B), 256, 0, stream>>>(
      o4p, nullptr, N4, nullptr, nullptr, 0, dtw, W(29), W(30), dto, nullptr, N4, 1024);
  // d4: concat(dt, o4p) 2048 -> 512 @ N4
  conv1x1_wmma<1024, 1024, false><<<dim3(512 / 16, 1, B), 256, 0, stream>>>(
      dto, nullptr, N4, o4p, nullptr, N4, d4w, W(32), W(33), d4o, nullptr, N4, 512);
  // d3: concat(interp_up3(d4o), o3p) 1024 -> 256 @ N3
  conv1x1_wmma<512, 512, false><<<dim3(256 / 16, (N3 + 127) / 128, B), 256, 0, stream>>>(
      d4o, up3, N4, o3p, nullptr, N3, d3w, W(35), W(36), d3o, nullptr, N3, 256);
  // d2: concat(interp_up2(d3o), o2p) 512 -> 128 @ N2
  conv1x1_wmma<256, 256, false><<<dim3(128 / 16, N2 / 128, B), 256, 0, stream>>>(
      d3o, up2, N3, o2p, nullptr, N2, d2w, W(38), W(39), d2o, nullptr, N2, 128);
  // d1: concat(interp_up1(d2o), o1p) 256 -> 64 @ N1
  conv1x1_wmma<128, 128, false><<<dim3(64 / 16, N1 / 128, B), 256, 0, stream>>>(
      d2o, up1, N2, o1p, nullptr, N1, d1w, W(41), W(42), d1o, nullptr, N1, 64);
  // d0: interp_up0(d1o) 64 -> 64 @ N0, f32 channel-major straight to d_out
  conv1x1_wmma<64, 0, true><<<dim3(64 / 16, N0 / 128, B), 256, 0, stream>>>(
      d1o, up0, N1, nullptr, nullptr, 0, d0w, W(44), W(45), nullptr, (float*)d_out, N0, 64);
}
